// SelfAttentionInteraction_15796889715496
// MI455X (gfx1250) — matrix operand order repaired and checked by reading, not compiled, not run on hardware
//
#include <hip/hip_runtime.h>

typedef __attribute__((ext_vector_type(16))) _Float16 v16h;
typedef __attribute__((ext_vector_type(8)))  _Float16 v8h;
typedef __attribute__((ext_vector_type(4)))  _Float16 v4h;
typedef __attribute__((ext_vector_type(8)))  float    v8f;
typedef __attribute__((ext_vector_type(4)))  unsigned int v4u;
typedef __attribute__((ext_vector_type(8)))  int v8i;
typedef __attribute__((ext_vector_type(4)))  int v4i;

#define HEADS 2
#define DH    8
#define FDIM  39
#define FPAD  48
#define EDIM  16
#define DDIM  16

// f16 staging region offsets (in halfs). All operand rows are 16B-aligned.
#define XA_OFF 0            // [48][32]  X,  A-layout rows (K=16 padded to 32)
#define WB_OFF 1536         // [4][16][32]  W^T per weight, B-layout rows
#define QA_OFF 3584         // [2][48][32]  Q per head, A-layout rows (K=8 pad 32)
#define KB_OFF 6656         // [2][48][32]  K^T per head, B-layout rows
#define VB_OFF 9728         // [2][16][64]  V^T per head, B-layout rows (K=48 pad 64)
#define PA_OFF 11776        // [2][48][64]  probs per head, A-layout rows
#define HTOT   17920

#if __has_builtin(__builtin_amdgcn_tensor_load_to_lds) && __has_builtin(__builtin_amdgcn_s_wait_tensorcnt)
#define USE_TDM 1
#else
#define USE_TDM 0
#endif

// A-operand: lane l holds row m; element K-order {0..7,16..23}(+8 for hi-half)
__device__ __forceinline__ v16h load_a(const _Float16* row, int hi) {
    v8h h0 = *(const v8h*)(row + 8 * hi);
    v8h h1 = *(const v8h*)(row + 16 + 8 * hi);
    return __builtin_shufflevector(h0, h1, 0,1,2,3,4,5,6,7,8,9,10,11,12,13,14,15);
}
// B-operand from B^T rows: lane l = column n; element K-order {0..7,16..23}(+8 for hi-half)
__device__ __forceinline__ v16h load_b(const _Float16* rowT, int hi) {
    v8h h0 = *(const v8h*)(rowT + 8 * hi);
    v8h h1 = *(const v8h*)(rowT + 16 + 8 * hi);
    return __builtin_shufflevector(h0, h1, 0,1,2,3,4,5,6,7,8,9,10,11,12,13,14,15);
}

__global__ __launch_bounds__(32)
void attn_ff_kernel(const float* __restrict__ x,
                    const float* __restrict__ Wq,
                    const float* __restrict__ Wk,
                    const float* __restrict__ Wv,
                    const float* __restrict__ Wr,
                    float* __restrict__ out)
{
    __shared__ __align__(16) _Float16 sH[HTOT];
    __shared__ __align__(16) float sS[HEADS * FPAD * FPAD]; // scores f32
    __shared__ __align__(16) float sO[FPAD * DDIM];         // residual + attn out
    __shared__ __align__(16) float sXf32[FDIM * EDIM];      // TDM landing buffer
    __shared__ float sInv[HEADS * FPAD];                    // 1/rowsum

    const int lane = threadIdx.x;
    const int hi   = lane >> 4;
    const int l    = lane & 15;
    const int b    = blockIdx.x;
    const float* xb = x + (size_t)b * FDIM * EDIM;

#if USE_TDM
    // ---- issue TDM: contiguous 624-float tile, global -> LDS (async) ----
    {
        unsigned long long ga = (unsigned long long)xb;
        unsigned int lds_off = (unsigned int)(unsigned long long)(void*)sXf32;
        // D# group0: count=1 | lds_addr | global_addr | type=2 ("image")
        v4u g0 = { 1u, lds_off, (unsigned int)ga,
                   (unsigned int)((ga >> 32) & 0x1FFFFFFu) | 0x80000000u };
        // D# group1: data_size=4B; tensor_dim0=tile_dim0=624; dim1=tile_dim1=1
        v8i g1 = { (int)(2u << 16),            // data_size=2 (4 bytes)
                   (int)(624u << 16),          // tensor_dim0[15:0] @ bits[63:48]
                   (int)(1u << 16),            // tensor_dim1[15:0] @ bits[95:80]
                   (int)(624u << 16),          // tile_dim0 @ bits[127:112]
                   (int)1,                     // tile_dim1 @ bits[143:128]
                   (int)624,                   // tensor_dim0_stride[31:0]
                   (int)(624u << 16),          // tensor_dim1_stride[15:0]
                   0 };
        v4i g2 = { 0, 0, 0, 0 };
        v4i g3 = { 0, 0, 0, 0 };
#if defined(__clang_major__) && (__clang_major__ >= 23)
        v8i g4 = { 0, 0, 0, 0, 0, 0, 0, 0 };
        __builtin_amdgcn_tensor_load_to_lds(g0, g1, g2, g3, g4, 0);
#else
        __builtin_amdgcn_tensor_load_to_lds(g0, g1, g2, g3, 0);
#endif
    }
#endif

    // ---- zero all f16 staging (covers every pad region); overlaps the DMA ----
    {
        v8h z = {};
        v8h* p = (v8h*)sH;
        for (int i = lane; i < HTOT / 8; i += 32) p[i] = z;
    }
    // ---- stage weights transposed into B-layout rows: WB[w][n][k] = W[k][n] ----
    for (int p = lane; p < 4 * DDIM; p += 32) {
        int w = p >> 4, n = p & 15;
        const float* Wsrc = (w == 0) ? Wq : (w == 1) ? Wk : (w == 2) ? Wv : Wr;
        v8h h0, h1;
        #pragma unroll
        for (int k = 0; k < 8; ++k)  h0[k] = (_Float16)Wsrc[k * DDIM + n];
        #pragma unroll
        for (int k = 0; k < 8; ++k)  h1[k] = (_Float16)Wsrc[(k + 8) * DDIM + n];
        *(v8h*)(sH + WB_OFF + p * 32)     = h0;
        *(v8h*)(sH + WB_OFF + p * 32 + 8) = h1;
    }
#if USE_TDM
    __builtin_amdgcn_s_wait_tensorcnt(0);
#endif
    __syncthreads();

    // ---- convert x tile f32 -> f16 A-layout rows (K padded to 32) ----
    {
#if USE_TDM
        const float4* src4 = (const float4*)sXf32;
#else
        const float4* src4 = (const float4*)xb;
#endif
        for (int i4 = lane; i4 < (FDIM * EDIM) / 4; i4 += 32) {
            float4 f = src4[i4];
            int base = i4 * 4, row = base / EDIM, k = base % EDIM;
            v4h h = { (_Float16)f.x, (_Float16)f.y, (_Float16)f.z, (_Float16)f.w };
            *(v4h*)(sH + XA_OFF + row * 32 + k) = h;
        }
    }
    __syncthreads();

    // ---- projections: Q,K,V scattered into WMMA-operand layouts; Res -> sO ----
    #pragma unroll
    for (int w = 0; w < 4; ++w) {
        #pragma unroll
        for (int t = 0; t < 3; ++t) {
            v16h a  = load_a(sH + XA_OFF + (16 * t + l) * 32, hi);
            v16h bm = load_b(sH + WB_OFF + w * 512 + l * 32, hi);
            v8f c = {};
            c = __builtin_amdgcn_wmma_f32_16x16x32_f16(false, a, false, bm,
                                                       (short)0, c, false, false);
            if (w <= 1) {          // Q -> A-layout rows ; K -> B^T-layout rows
                int base = (w == 0 ? QA_OFF : KB_OFF) + (l >> 3) * 1536 + (l & 7);
                #pragma unroll
                for (int r = 0; r < 8; ++r)
                    sH[base + (16 * t + r + 8 * hi) * 32] = (_Float16)c[r];
            } else if (w == 2) {   // V^T: VB[h][dcol][g]
                int base = VB_OFF + (l >> 3) * 1024 + (l & 7) * 64;
                #pragma unroll
                for (int r = 0; r < 8; ++r)
                    sH[base + 16 * t + r + 8 * hi] = (_Float16)c[r];
            } else {               // residual projection, f32
                #pragma unroll
                for (int r = 0; r < 8; ++r)
                    sO[(16 * t + r + 8 * hi) * DDIM + l] = c[r];
            }
        }
    }
    __syncthreads();

    // ---- scores: S_h = Q_h @ K_h^T ----
    #pragma unroll
    for (int h = 0; h < HEADS; ++h) {
        #pragma unroll
        for (int tq = 0; tq < 3; ++tq) {
            #pragma unroll
            for (int tk = 0; tk < 3; ++tk) {
                v16h a  = load_a(sH + QA_OFF + h * 1536 + (16 * tq + l) * 32, hi);
                v16h bm = load_b(sH + KB_OFF + h * 1536 + (16 * tk + l) * 32, hi);
                v8f c = {};
                c = __builtin_amdgcn_wmma_f32_16x16x32_f16(false, a, false, bm,
                                                           (short)0, c, false, false);
                #pragma unroll
                for (int r = 0; r < 8; ++r)
                    sS[h * (FPAD * FPAD) + (16 * tq + r + 8 * hi) * FPAD + 16 * tk + l] = c[r];
            }
        }
    }
    __syncthreads();

    // ---- softmax, vectorized with compile-time masks; stores UNNORMALIZED
    //      exp probs (f16) + 1/sum (normalization folded into PV store) ----
    for (int idx = lane; idx < HEADS * FPAD; idx += 32) {
        int h = idx / FPAD, f = idx % FPAD;
        const float4* s4 = (const float4*)&sS[h * (FPAD * FPAD) + f * FPAD];
        _Float16*    arow = sH + PA_OFF + h * 3072 + f * 64;
        float m = -3.0e38f;
        #pragma unroll
        for (int j = 0; j < 10; ++j) {              // cols 0..39, mask >=39
            float4 v = s4[j];
            m = fmaxf(m, v.x);
            if (4 * j + 1 < FDIM) m = fmaxf(m, v.y);
            if (4 * j + 2 < FDIM) m = fmaxf(m, v.z);
            if (4 * j + 3 < FDIM) m = fmaxf(m, v.w);
        }
        float s = 0.f;
        #pragma unroll
        for (int j = 0; j < 5; ++j) {               // 8 cols per step -> one b128 store
            float4 va = s4[2 * j], vb = s4[2 * j + 1];
            float vv[8] = { va.x, va.y, va.z, va.w, vb.x, vb.y, vb.z, vb.w };
            v8h hch;
            #pragma unroll
            for (int c = 0; c < 8; ++c) {
                int g = 8 * j + c;
                float ev = (g < FDIM) ? __expf(vv[c] - m) : 0.f;
                s += ev;
                hch[c] = (_Float16)ev;
            }
            *(v8h*)(arow + 8 * j) = hch;
        }
        sInv[idx] = 1.f / s;
    }
    __syncthreads();

    // ---- out: O_h = P_h @ V_h, K=48 padded to 64 -> two K=32 steps ----
    #pragma unroll
    for (int h = 0; h < HEADS; ++h) {
        #pragma unroll
        for (int tf = 0; tf < 3; ++tf) {
            v8f c = {};
            #pragma unroll
            for (int ks = 0; ks < 2; ++ks) {
                v16h a  = load_a(sH + PA_OFF + h * 3072 + (16 * tf + l) * 64 + 32 * ks, hi);
                v16h bm = load_b(sH + VB_OFF + h * 1024 + l * 64 + 32 * ks, hi);
                c = __builtin_amdgcn_wmma_f32_16x16x32_f16(false, a, false, bm,
                                                           (short)0, c, false, false);
            }
            if (l < DH) {
                #pragma unroll
                for (int r = 0; r < 8; ++r) {
                    int row = 16 * tf + r + 8 * hi;
                    sO[row * DDIM + h * DH + l] += c[r] * sInv[h * FPAD + row];
                }
            }
        }
    }
    __syncthreads();

    // ---- ReLU + vectorized store ----
    {
        float4* out4 = (float4*)(out + (size_t)b * FDIM * EDIM);
        const float4* so4 = (const float4*)sO;
        for (int i4 = lane; i4 < (FDIM * EDIM) / 4; i4 += 32) {
            float4 v = so4[i4];
            v.x = v.x > 0.f ? v.x : 0.f;
            v.y = v.y > 0.f ? v.y : 0.f;
            v.z = v.z > 0.f ? v.z : 0.f;
            v.w = v.w > 0.f ? v.w : 0.f;
            out4[i4] = v;
        }
    }
}

extern "C" void kernel_launch(void* const* d_in, const int* in_sizes, int n_in,
                              void* d_out, int out_size, void* d_ws, size_t ws_size,
                              hipStream_t stream) {
    const float* x  = (const float*)d_in[0];
    const float* Wq = (const float*)d_in[1];
    const float* Wk = (const float*)d_in[2];
    const float* Wv = (const float*)d_in[3];
    const float* Wr = (const float*)d_in[4];
    float* out = (float*)d_out;
    int B = in_sizes[0] / (FDIM * EDIM);   // 16384
    attn_ff_kernel<<<B, 32, 0, stream>>>(x, Wq, Wk, Wv, Wr, out);
}
